// Speller_65369402245757
// MI455X (gfx1250) — matrix operand order repaired and checked by reading, not compiled, not run on hardware
//
#include <hip/hip_runtime.h>
#include <hip/hip_bf16.h>
#include <math.h>

// ---------------------------------------------------------------------------
// Speller (LAS decoder) on MI455X, cluster-cooperative version.
//
//   grid = 128 WGs = 8 clusters x 16 members (dispatch with {16,1,1} clusters;
//   falls back to an L2 atomic barrier when ClusterID==0 / plain launch).
//
//   member 0  : WMMA bf16 LSTM1+LSTM2 (M=16 batch tile), gates fused in VGPRs,
//               FC logits. Weights are L2-resident (9MB bf16) -> L2-bound GEMMs.
//   members 1-15 : one batch row of attention each (scores -> softmax -> ctx),
//               streaming key/value from HBM. 120 WGPs stream concurrently,
//               which is what the 23.3 TB/s HBM roofline actually requires.
//
//   Per-step handoff: h1 (16x512 f32) and ctx (16x512 f32) via L2 scratch,
//   ordered by the cluster split-barrier (s_barrier_signal/-wait -3).
// ---------------------------------------------------------------------------

typedef __attribute__((ext_vector_type(16))) __bf16 bf16x16;
typedef __attribute__((ext_vector_type(8)))  float  f32x8;

#define Bsz 128
#define Ssz 800
#define Hsz 512
#define Esz 256
#define Vsz 34
#define Tsz 128
#define XK  768            // E + H (LSTM1 input width, weight K stride)
#define OUTW (Tsz * Vsz)

// padded LDS row strides (stride % 64 banks == 4 dwords -> conflict-free
// b128 A-fragment loads across the 16 M-rows)
#define HP  520            // 512 + 8 bf16
#define XKP 776            // 768 + 8 bf16

// ---- compute-WG LDS layout (bytes) ----------------------------------------
#define OFF_C0     0                    // 16 x HP f32 = 33280
#define OFF_C1     33280
#define OFF_H1F    66560
#define OFF_XBF    99840                // 16 x XKP bf16 = 24832
#define OFF_H0BF   124672               // 16 x HP bf16 = 16640
#define OFF_H1BF   141312
#define SMEM_BYTES 157952
// ---- attention-member LDS layout (aliases the same allocation) -------------
#define OFF_MH1    0                    // 512 f32
#define OFF_ESC    2048                 // 800 f32
#define OFF_CTXA   5248                 // 512 f32
#define OFF_WRED   7296                 // 32 f32

// ---- workspace layout (bytes) ----------------------------------------------
#define N1 (2048ull * 768)              // Wih1 elements
#define N2 (2048ull * 512)              // Whh1/Wih2/Whh2 elements
#define SCR_H1   9437184ull             // 8 tiles x 16 x 512 f32 = 262144 B
#define SCR_CTX  9699328ull             // same size
#define SCR_BAR  9961472ull             // 8 x u32 arrive counters

__device__ __forceinline__ float sigf(float x) { return 1.0f / (1.0f + __expf(-x)); }

__device__ __forceinline__ void cluster_signal() {
    asm volatile("s_barrier_signal -3" ::: "memory");
}
__device__ __forceinline__ void cluster_wait() {
    asm volatile("s_barrier_wait -3" ::: "memory");
}
__device__ __forceinline__ void bar_arrive(unsigned* bar) {
    __threadfence();
    __hip_atomic_fetch_add(bar, 1u, __ATOMIC_ACQ_REL, __HIP_MEMORY_SCOPE_AGENT);
}
__device__ __forceinline__ void bar_spin(unsigned* bar, unsigned target) {
    while (__hip_atomic_load(bar, __ATOMIC_ACQUIRE, __HIP_MEMORY_SCOPE_AGENT) < target)
        __builtin_amdgcn_s_sleep(8);
}
// full barrier round (sync + signal + wait), used where no overlap is needed
__device__ __forceinline__ void cbar(int incluster, unsigned* bar, int round,
                                     int wave) {
    __syncthreads();
    if (incluster) {
        if (wave == 0) cluster_signal();
        cluster_wait();
    } else {
        if (threadIdx.x == 0) { bar_arrive(bar); bar_spin(bar, 16u * round); }
        __syncthreads();
    }
}

// A fragment (16x32 bf16 MxK) from padded LDS, contiguous k = 16*(lane/16) + s, same order as B.
__device__ __forceinline__ bf16x16 load_a_frag(const __bf16* X, int ldk,
                                               int m, int kq, int k0) {
    bf16x16 a;
    const __bf16* row = X + m * ldk + k0 + kq;
#pragma unroll
    for (int j = 0; j < 8; ++j) {
        int kk = 2 * j;
        a[2 * j]     = row[kk];
        a[2 * j + 1] = row[kk + 1];
    }
    return a;
}

__global__ void prep_weights_bf16(const float* __restrict__ Wih1,
                                  const float* __restrict__ Whh1,
                                  const float* __restrict__ Wih2,
                                  const float* __restrict__ Whh2,
                                  __bf16* __restrict__ dst,
                                  unsigned* __restrict__ bar) {
    if (blockIdx.x == 0 && threadIdx.x < 8) bar[threadIdx.x] = 0u;  // reset barriers
    size_t idx = (size_t)blockIdx.x * 256u + threadIdx.x;
    if (idx < N1)            { dst[idx] = (__bf16)Wih1[idx]; return; }
    idx -= N1;
    if (idx < N2)            { dst[N1 + idx] = (__bf16)Whh1[idx]; return; }
    idx -= N2;
    if (idx < N2)            { dst[N1 + N2 + idx] = (__bf16)Wih2[idx]; return; }
    idx -= N2;
    if (idx < N2)            { dst[N1 + 2 * N2 + idx] = (__bf16)Whh2[idx]; return; }
}

__global__ __launch_bounds__(512) void speller_main(
    const float* __restrict__ key, const float* __restrict__ value,
    const int* __restrict__ decode, const int* __restrict__ lens,
    const float* __restrict__ emb,
    const __bf16* __restrict__ Wih1, const __bf16* __restrict__ Whh1,
    const float* __restrict__ b1,
    const __bf16* __restrict__ Wih2, const __bf16* __restrict__ Whh2,
    const float* __restrict__ b2,
    const float* __restrict__ fcW, const float* __restrict__ fcb,
    float* __restrict__ g_h1, float* __restrict__ g_ctx,
    unsigned* __restrict__ g_bar,
    float* __restrict__ out) {
    extern __shared__ char smem[];
    const int tid  = threadIdx.x;
    const int w    = tid >> 5;      // wave id (wave32)
    const int lane = tid & 31;
    const int tile = blockIdx.x >> 4;   // cluster / batch tile
    const int mem  = blockIdx.x & 15;   // workgroup-in-cluster
    const int b0   = tile * 16;
    const int incluster = (__builtin_amdgcn_cluster_id_x() != 0);
    unsigned* bar = g_bar + tile;
    int round = 0;

    if (mem == 0) {
        // ================== compute workgroup: LSTM GEMMs + FC =============
        float*  c0   = (float*)(smem + OFF_C0);
        float*  c1   = (float*)(smem + OFF_C1);
        float*  h1f  = (float*)(smem + OFF_H1F);
        __bf16* xbf  = (__bf16*)(smem + OFF_XBF);
        __bf16* h0bf = (__bf16*)(smem + OFF_H0BF);
        __bf16* h1bf = (__bf16*)(smem + OFF_H1BF);

        const int n_q   = lane & 15;
        const int kqA   = (lane >> 4) * 16;
        const int kqB   = (lane >> 4) * 16;
        const int mhalf = (lane >> 4) * 8;

        for (int i = tid; i < 16 * HP; i += 512) {
            c0[i] = 0.f; c1[i] = 0.f; h1f[i] = 0.f;
            h0bf[i] = (__bf16)0.f; h1bf[i] = (__bf16)0.f;
        }
        for (int i = tid; i < 16 * Hsz; i += 512) {  // ctx0 = key[:,0,:]
            int m = i >> 9, h = i & 511;
            xbf[m * XKP + h] = (__bf16)key[((size_t)(b0 + m) * Ssz) * Hsz + h];
        }
        __syncthreads();

        for (int t = 0; t < Tsz; ++t) {
            if (t > 0) {  // ingest ctx published by attention members
                __threadfence();
                for (int i = tid; i < 16 * Hsz; i += 512) {
                    int m = i >> 9, h = i & 511;
                    xbf[m * XKP + h] = (__bf16)g_ctx[(size_t)(b0 + m) * Hsz + h];
                }
            }
            for (int i = tid; i < 16 * Esz; i += 512) {  // embedding gather
                int m = i >> 8, e = i & 255;
                int tok = decode[(b0 + m) * Tsz + t];
                xbf[m * XKP + Hsz + e] = (__bf16)emb[tok * Esz + e];
            }
            __syncthreads();

            // ---- LSTM1: g = x@Wih1^T + h0@Whh1^T + b1
            {
                f32x8 acc[8];
#pragma unroll
                for (int i = 0; i < 8; ++i) {
                    float bv = b1[(w + 16 * i) * 16 + n_q];
#pragma unroll
                    for (int r = 0; r < 8; ++r) acc[i][r] = bv;
                }
                for (int k0 = 0; k0 < XK; k0 += 32) {
                    bf16x16 a = load_a_frag(xbf, XKP, n_q, kqA, k0);
#pragma unroll
                    for (int i = 0; i < 8; ++i) {
                        int n = (w + 16 * i) * 16 + n_q;
                        bf16x16 bfr = *(const bf16x16*)(Wih1 + (size_t)n * XK + k0 + kqB);
                        acc[i] = __builtin_amdgcn_wmma_f32_16x16x32_bf16(
                            false, a, false, bfr, (short)0, acc[i], false, false);
                    }
                }
                for (int k0 = 0; k0 < Hsz; k0 += 32) {
                    bf16x16 a = load_a_frag(h0bf, HP, n_q, kqA, k0);
#pragma unroll
                    for (int i = 0; i < 8; ++i) {
                        int n = (w + 16 * i) * 16 + n_q;
                        bf16x16 bfr = *(const bf16x16*)(Whh1 + (size_t)n * Hsz + k0 + kqB);
                        acc[i] = __builtin_amdgcn_wmma_f32_16x16x32_bf16(
                            false, a, false, bfr, (short)0, acc[i], false, false);
                    }
                }
                __syncthreads();  // all waves done reading h0bf
#pragma unroll
                for (int ii = 0; ii < 2; ++ii) {  // gates fused in VGPRs
                    int j = (w + 16 * ii) * 16 + n_q;
#pragma unroll
                    for (int r = 0; r < 8; ++r) {
                        int m = r + mhalf;
                        float iv = acc[ii][r],     fv = acc[ii + 2][r];
                        float gv = acc[ii + 4][r], ov = acc[ii + 6][r];
                        float cn = sigf(fv) * c0[m * HP + j] + sigf(iv) * tanhf(gv);
                        float hv = sigf(ov) * tanhf(cn);
                        c0[m * HP + j]   = cn;
                        h0bf[m * HP + j] = (__bf16)hv;
                    }
                }
            }
            __syncthreads();

            // ---- LSTM2: g = h0@Wih2^T + h1@Whh2^T + b2
            {
                f32x8 acc[8];
#pragma unroll
                for (int i = 0; i < 8; ++i) {
                    float bv = b2[(w + 16 * i) * 16 + n_q];
#pragma unroll
                    for (int r = 0; r < 8; ++r) acc[i][r] = bv;
                }
                for (int k0 = 0; k0 < Hsz; k0 += 32) {
                    bf16x16 a = load_a_frag(h0bf, HP, n_q, kqA, k0);
#pragma unroll
                    for (int i = 0; i < 8; ++i) {
                        int n = (w + 16 * i) * 16 + n_q;
                        bf16x16 bfr = *(const bf16x16*)(Wih2 + (size_t)n * Hsz + k0 + kqB);
                        acc[i] = __builtin_amdgcn_wmma_f32_16x16x32_bf16(
                            false, a, false, bfr, (short)0, acc[i], false, false);
                    }
                }
                for (int k0 = 0; k0 < Hsz; k0 += 32) {
                    bf16x16 a = load_a_frag(h1bf, HP, n_q, kqA, k0);
#pragma unroll
                    for (int i = 0; i < 8; ++i) {
                        int n = (w + 16 * i) * 16 + n_q;
                        bf16x16 bfr = *(const bf16x16*)(Whh2 + (size_t)n * Hsz + k0 + kqB);
                        acc[i] = __builtin_amdgcn_wmma_f32_16x16x32_bf16(
                            false, a, false, bfr, (short)0, acc[i], false, false);
                    }
                }
                __syncthreads();  // all waves done reading h1bf
#pragma unroll
                for (int ii = 0; ii < 2; ++ii) {
                    int j = (w + 16 * ii) * 16 + n_q;
#pragma unroll
                    for (int r = 0; r < 8; ++r) {
                        int m = r + mhalf;
                        float iv = acc[ii][r],     fv = acc[ii + 2][r];
                        float gv = acc[ii + 4][r], ov = acc[ii + 6][r];
                        float cn = sigf(fv) * c1[m * HP + j] + sigf(iv) * tanhf(gv);
                        float hv = sigf(ov) * tanhf(cn);
                        c1[m * HP + j]   = cn;
                        h1f[m * HP + j]  = hv;
                        h1bf[m * HP + j] = (__bf16)hv;
                        g_h1[(size_t)(b0 + m) * Hsz + j] = hv;  // publish h1
                    }
                }
            }
            __threadfence();
            __syncthreads();
            ++round;  // ---- barrier A (h1 published), FC overlapped with wait
            if (incluster) {
                if (w == 0) cluster_signal();
            } else if (tid == 0) {
                bar_arrive(bar);
            }
            for (int i = tid; i < 16 * Vsz; i += 512) {  // FC logits
                int m = i / Vsz, v = i % Vsz;
                float s = fcb[v];
                const float* hr = h1f + m * HP;
                const float* wr = fcW + v * Hsz;
                for (int k = 0; k < Hsz; ++k) s += hr[k] * wr[k];
                out[(size_t)(b0 + m) * OUTW + t * Vsz + v] = s;
            }
            if (incluster) {
                cluster_wait();
            } else {
                if (tid == 0) bar_spin(bar, 16u * round);
                __syncthreads();
            }
            ++round;  // ---- barrier B (ctx published by members)
            cbar(incluster, bar, round, w);
        }
    } else {
        // ================== attention members: one batch row each ==========
        float* h1row = (float*)(smem + OFF_MH1);
        float* esc   = (float*)(smem + OFF_ESC);
        float* ctxa  = (float*)(smem + OFF_CTXA);
        float* wred  = (float*)(smem + OFF_WRED);

        for (int t = 0; t < Tsz; ++t) {
            ++round;                       // barrier A: wait for h1
            cbar(incluster, bar, round, w);
            __threadfence();

            for (int r = mem - 1; r < 16; r += 15) {  // member 1 also row 15
                const int b = b0 + r;
                h1row[tid] = g_h1[(size_t)b * Hsz + tid];
                if (tid < Hsz) ctxa[tid] = 0.f;
                __syncthreads();

                // scores: e[s] = key[b,s,:].h1  (wave w -> s in [50w,50w+50))
                for (int q = 0; q < 50; ++q) {
                    int s = w * 50 + q;
                    const float* kr = key + ((size_t)b * Ssz + s) * Hsz;
                    float part = 0.f;
#pragma unroll 4
                    for (int h = lane; h < Hsz; h += 32) part += kr[h] * h1row[h];
                    for (int off = 16; off; off >>= 1)
                        part += __shfl_xor(part, off, 32);
                    if (lane == 0) esc[s] = part;
                }
                __syncthreads();

                // softmax over full S, then mask (s >= len -> 1e-9)
                float mx = -1e30f;
                for (int s = tid; s < Ssz; s += 512) mx = fmaxf(mx, esc[s]);
                for (int off = 16; off; off >>= 1)
                    mx = fmaxf(mx, __shfl_xor(mx, off, 32));
                if (lane == 0) wred[w] = mx;
                __syncthreads();
                if (w == 0) {
                    float v = (lane < 16) ? wred[lane] : -1e30f;
                    for (int off = 16; off; off >>= 1)
                        v = fmaxf(v, __shfl_xor(v, off, 32));
                    if (lane == 0) wred[0] = v;
                }
                __syncthreads();
                mx = wred[0];
                float sum = 0.f;
                for (int s = tid; s < Ssz; s += 512) {
                    float e = __expf(esc[s] - mx);
                    esc[s] = e;
                    sum += e;
                }
                for (int off = 16; off; off >>= 1) sum += __shfl_xor(sum, off, 32);
                if (lane == 0) wred[16 + w] = sum;
                __syncthreads();
                if (w == 0) {
                    float v = (lane < 16) ? wred[16 + lane] : 0.f;
                    for (int off = 16; off; off >>= 1) v += __shfl_xor(v, off, 32);
                    if (lane == 0) wred[16] = v;
                }
                __syncthreads();
                float inv = 1.0f / wred[16];
                int L = lens[b];
                for (int s = tid; s < Ssz; s += 512) {
                    float e = esc[s] * inv;
                    esc[s] = (s >= L) ? 1e-9f : e;
                }
                __syncthreads();

                // ctx[h] = sum_s value[b,s,h]*e[s]; per-wave regs + LDS atomics
                {
                    float a[16];
#pragma unroll
                    for (int j = 0; j < 16; ++j) a[j] = 0.f;
                    const float* vb = value + (size_t)b * Ssz * Hsz;
                    for (int q = 0; q < 50; ++q) {
                        int s = w * 50 + q;
                        float e = esc[s];
                        const float* vr = vb + (size_t)s * Hsz;
#pragma unroll
                        for (int j = 0; j < 16; ++j) a[j] += e * vr[lane + 32 * j];
                    }
#pragma unroll
                    for (int j = 0; j < 16; ++j)
                        atomicAdd(&ctxa[lane + 32 * j], a[j]);  // ds_add_f32
                }
                __syncthreads();
                g_ctx[(size_t)b * Hsz + tid] = ctxa[tid];  // publish ctx row
                __syncthreads();
            }
            __threadfence();
            ++round;                       // barrier B: ctx published
            cbar(incluster, bar, round, w);
        }
    }
}

extern "C" void kernel_launch(void* const* d_in, const int* in_sizes, int n_in,
                              void* d_out, int out_size, void* d_ws, size_t ws_size,
                              hipStream_t stream) {
    const float* key    = (const float*)d_in[0];
    const float* value  = (const float*)d_in[1];
    const int*   decode = (const int*)d_in[2];
    const int*   lens   = (const int*)d_in[3];
    const float* emb    = (const float*)d_in[4];
    const float* Wih1f  = (const float*)d_in[5];
    const float* Whh1f  = (const float*)d_in[6];
    const float* b1     = (const float*)d_in[7];
    const float* Wih2f  = (const float*)d_in[8];
    const float* Whh2f  = (const float*)d_in[9];
    const float* b2     = (const float*)d_in[10];
    const float* fcW    = (const float*)d_in[11];
    const float* fcb    = (const float*)d_in[12];

    char* ws = (char*)d_ws;
    __bf16*   wsb   = (__bf16*)ws;
    float*    g_h1  = (float*)(ws + SCR_H1);
    float*    g_ctx = (float*)(ws + SCR_CTX);
    unsigned* g_bar = (unsigned*)(ws + SCR_BAR);

    const size_t total = N1 + 3 * N2;
    prep_weights_bf16<<<(unsigned)((total + 255) / 256), 256, 0, stream>>>(
        Wih1f, Whh1f, Wih2f, Whh2f, wsb, g_bar);

    hipFuncSetAttribute((const void*)speller_main,
                        hipFuncAttributeMaxDynamicSharedMemorySize, SMEM_BYTES);
    // 8 clusters x 16 member WGs (blockIdx: tile = bx>>4, member = bx&15)
    speller_main<<<Bsz, 512, SMEM_BYTES, stream>>>(
        key, value, decode, lens, emb,
        wsb, wsb + N1, b1, wsb + N1 + N2, wsb + N1 + 2 * N2, b2,
        fcW, fcb, g_h1, g_ctx, g_bar, (float*)d_out);
}
